// VisionTransformer_34840774705286
// MI455X (gfx1250) — matrix operand order repaired and checked by reading, not compiled, not run on hardware
//
#include <hip/hip_runtime.h>

// ---------------- model constants ----------------
#define B_    16
#define C_    768
#define NH_   12
#define HD_   64
#define N_    196
#define R_    49
#define NQ_   245      // N_ + R_
#define HP_   14
#define DEPTH_ 12
#define FF_   3072
#define NCLS_ 1000

typedef __attribute__((ext_vector_type(16))) __bf16        v16bf;
typedef __attribute__((ext_vector_type(2)))  __bf16        v2bf;
typedef __attribute__((ext_vector_type(8)))  float         v8f;
typedef __attribute__((ext_vector_type(8)))  unsigned int  v8u;

// hardware bf16 convert (v_cvt_pk_bf16_f32), RNE
__device__ __forceinline__ unsigned int pack2(float a, float b) {
  v2bf p;
  p.x = (__bf16)a;
  p.y = (__bf16)b;
  return __builtin_bit_cast(unsigned int, p);
}
__device__ __forceinline__ unsigned short f2bf(float f) {
  return __builtin_bit_cast(unsigned short, (__bf16)f);
}
__device__ __forceinline__ v16bf frag_cast(v8u w) { return __builtin_bit_cast(v16bf, w); }
__device__ __forceinline__ v8u cat4(uint4 lo, uint4 hi) {
  v8u w = {lo.x, lo.y, lo.z, lo.w, hi.x, hi.y, hi.z, hi.w};
  return w;
}

// =====================================================================
// Generic batched GEMM:  out[b] = act(A[b] @ W + bias) (+ resid[b])
// Tile 128x128x32, statically double-buffered LDS, 8 waves x (32x64).
// K must be a multiple of 64 (true: 768 / 3072).
// Out-of-range rows/cols CLAMPED during staging (outputs never stored).
// Staging addresses are per-thread pointers post-incremented by the
// K-step stride (no per-iteration 64-bit multiplies).
// LDS row stride = (BK+LPAD)*2 = 96B -> fragments are 2x aligned b128 DS loads.
// =====================================================================
#define BM 128
#define BN 128
#define BK 32
#define LPAD 16

__device__ __forceinline__ void g_load(const float2** ap, const float** wp0, const float** wp1,
                                       long stepw, float2* av, float2* wv)
{
  #pragma unroll
  for (int i = 0; i < 8; ++i) {
    av[i] = *ap[i];
    ap[i] += BK / 2;               // advance one K-tile (BK floats)
  }
  #pragma unroll
  for (int i = 0; i < 8; ++i) {
    wv[i].x = *wp0[i];
    wv[i].y = *wp1[i];
    wp0[i] += stepw;
    wp1[i] += stepw;
  }
}

__device__ __forceinline__ void l_store(unsigned short (* __restrict__ As)[BK + LPAD],
                                        unsigned short (* __restrict__ Ws)[BK + LPAD],
                                        int t, const float2* av, const float2* wv)
{
  #pragma unroll
  for (int i = 0; i < 8; ++i) {
    int e = i * 256 + t;
    int r = e >> 4, k = (e & 15) * 2;
    *(unsigned int*)&As[r][k] = pack2(av[i].x, av[i].y);
  }
  #pragma unroll
  for (int i = 0; i < 8; ++i) {
    int e = i * 256 + t;
    int n = e & 127, k = (e >> 7) * 2;
    *(unsigned int*)&Ws[n][k] = pack2(wv[i].x, wv[i].y);
  }
}

__device__ __forceinline__ void mma_step(const unsigned short (* __restrict__ As)[BK + LPAD],
                                         const unsigned short (* __restrict__ Ws)[BK + LPAD],
                                         int wm, int wn, int hsel, int l16, v8f acc[2][4])
{
  v16bf af[2], bfr[4];
  #pragma unroll
  for (int mi = 0; mi < 2; ++mi) {
    int row = wm * 32 + mi * 16 + l16;
    uint4 lo = *(const uint4*)&As[row][hsel * 16];       // kk = hsel*16 + 2v (contiguous)
    uint4 hi = *(const uint4*)&As[row][hsel * 16 + 8];
    af[mi] = frag_cast(cat4(lo, hi));
  }
  #pragma unroll
  for (int ni = 0; ni < 4; ++ni) {
    int col = wn * 64 + ni * 16 + l16;
    uint4 lo = *(const uint4*)&Ws[col][hsel * 16];       // kk = hsel*16 + 2v (contiguous)
    uint4 hi = *(const uint4*)&Ws[col][hsel * 16 + 8];
    bfr[ni] = frag_cast(cat4(lo, hi));
  }
  #pragma unroll
  for (int mi = 0; mi < 2; ++mi)
    #pragma unroll
    for (int ni = 0; ni < 4; ++ni)
      acc[mi][ni] = __builtin_amdgcn_wmma_f32_16x16x32_bf16(
          false, af[mi], false, bfr[ni], (short)0, acc[mi][ni], false, false);
}

__global__ __launch_bounds__(256) void gemm_kernel(
    const float* __restrict__ A, const float* __restrict__ W,
    const float* __restrict__ bias, const float* __restrict__ resid,
    float* __restrict__ out,
    int M, int K, int N, long sA, long sC, long sR, int act)
{
  __shared__ __attribute__((aligned(16))) unsigned short As[2][BM][BK + LPAD];
  __shared__ __attribute__((aligned(16))) unsigned short Ws[2][BN][BK + LPAD];

  const int t    = threadIdx.x;
  const int lane = t & 31;
  const int wave = t >> 5;
  const int wm   = wave >> 1;                 // 0..3  -> row group *32
  const int wn   = wave & 1;                  // 0..1  -> col group *64
  const int hsel = (lane >> 4) & 1;           // lane half
  const int l16  = lane & 15;
  const long bz  = blockIdx.z;
  const float* Ab = A + bz * sA;
  float*       Cb = out + bz * sC;
  const int m0 = blockIdx.x * BM;
  const int n0 = blockIdx.y * BN;
  const int nk = K / BK;                      // even (K % 64 == 0)

  // per-thread staging pointers (post-incremented per K-tile)
  const float2* ap[8];
  const float*  wp0[8];
  const float*  wp1[8];
  #pragma unroll
  for (int i = 0; i < 8; ++i) {
    int e = i * 256 + t;
    int r = e >> 4, k = (e & 15) * 2;
    int gr = m0 + r; gr = gr < M ? gr : M - 1;
    ap[i] = (const float2*)&Ab[(long)gr * K + k];
  }
  #pragma unroll
  for (int i = 0; i < 8; ++i) {
    int e = i * 256 + t;
    int n = e & 127, k = (e >> 7) * 2;
    int gn = n0 + n; gn = gn < N ? gn : N - 1;
    wp0[i] = W + (long)k * N + gn;
    wp1[i] = wp0[i] + N;
  }
  const long stepw = (long)BK * N;

  const v8f vzero = {0.f,0.f,0.f,0.f,0.f,0.f,0.f,0.f};
  v8f acc[2][4];
  #pragma unroll
  for (int i = 0; i < 2; ++i)
    #pragma unroll
    for (int j = 0; j < 4; ++j) acc[i][j] = vzero;

  float2 av[8], wv[8];

  g_load(ap, wp0, wp1, stepw, av, wv);        // tile 0
  l_store(As[0], Ws[0], t, av, wv);
  __syncthreads();

  for (int it = 0; it < nk; it += 2) {
    // even step: compute buf0, stage tile it+1 into buf1
    g_load(ap, wp0, wp1, stepw, av, wv);
    if (it + 2 < nk) {   // prefetch tile it+2 (pointers already advanced there)
      __builtin_prefetch((const float*)ap[0], 0, 1);
      __builtin_prefetch(wp0[0], 0, 1);
    }
    mma_step(As[0], Ws[0], wm, wn, hsel, l16, acc);
    l_store(As[1], Ws[1], t, av, wv);
    __syncthreads();

    // odd step: compute buf1, stage tile it+2 into buf0
    const bool more = (it + 2) < nk;
    if (more) g_load(ap, wp0, wp1, stepw, av, wv);
    mma_step(As[1], Ws[1], wm, wn, hsel, l16, acc);
    if (more) l_store(As[0], Ws[0], t, av, wv);
    __syncthreads();
  }

  // ---- epilogue: bias, activation, residual ----
  const bool full = (m0 + BM <= M) && (n0 + BN <= N);
  #pragma unroll
  for (int mi = 0; mi < 2; ++mi) {
    #pragma unroll
    for (int ni = 0; ni < 4; ++ni) {
      int col = n0 + wn * 64 + ni * 16 + l16;
      #pragma unroll
      for (int r = 0; r < 8; ++r) {
        int row = m0 + wm * 32 + mi * 16 + hsel * 8 + r;   // ISA C layout
        if (full || (row < M && col < N)) {
          float v = acc[mi][ni][r];
          if (bias)  v += bias[col];
          if (act == 1) v = 0.5f * v * (1.0f + erff(v * 0.70710678118654752f));
          if (resid) v += resid[bz * sR + (long)row * N + col];
          Cb[(long)row * N + col] = v;
        }
      }
    }
  }
}

// =====================================================================
// Attention: one block per (b, head, qtile-chunk); 8 waves, one 16-row
// q tile each. QK^T + softmax (registers) + P@V, all WMMA bf16.
// =====================================================================
__global__ __launch_bounds__(256) void attn_kernel(
    const float* __restrict__ q, const float* __restrict__ kv, float* __restrict__ o)
{
  __shared__ __attribute__((aligned(16))) unsigned short Vt[HD_][224];     // [d][key]
  __shared__ __attribute__((aligned(16))) unsigned short Pbuf[8][16 * 32]; // per-wave P

  const int b = blockIdx.x / NH_;
  const int h = blockIdx.x % NH_;
  const int t = threadIdx.x, lane = t & 31, wave = t >> 5;
  const int hsel = (lane >> 4) & 1, l16 = lane & 15;
  const float* kvb = kv + (long)b * N_ * (2 * C_);

  // stage V transposed; padded keys clamped (they multiply p = 0 later)
  for (int e = t; e < HD_ * 112; e += 256) {
    int d = e / 112, tok = (e % 112) * 2;
    int t0c = tok     < N_ ? tok     : N_ - 1;
    int t1c = tok + 1 < N_ ? tok + 1 : N_ - 1;
    float v0 = kvb[(long)t0c * (2 * C_) + C_ + h * HD_ + d];
    float v1 = kvb[(long)t1c * (2 * C_) + C_ + h * HD_ + d];
    *(unsigned int*)&Vt[d][tok] = pack2(v0, v1);
  }
  __syncthreads();

  const int qtile = blockIdx.y * 8 + wave;     // 0..15
  const int qrow  = qtile * 16 + l16;
  const int qr    = qrow < NQ_ ? qrow : NQ_ - 1;
  const float* qp = q + ((long)b * NQ_ + qr) * C_ + h * HD_;

  // Q A-fragments (float4 global loads; lanes hold contiguous 4-float runs)
  v16bf qa[2];
  #pragma unroll
  for (int kc = 0; kc < 2; ++kc) {
    const float* base = qp + kc * 32 + hsel * 16;
    float4 f0 = *(const float4*)(base + 0);
    float4 f1 = *(const float4*)(base + 4);
    float4 f2 = *(const float4*)(base + 8);
    float4 f3 = *(const float4*)(base + 12);
    v8u w = { pack2(f0.x, f0.y), pack2(f0.z, f0.w),
              pack2(f1.x, f1.y), pack2(f1.z, f1.w),
              pack2(f2.x, f2.y), pack2(f2.z, f2.w),
              pack2(f3.x, f3.y), pack2(f3.z, f3.w) };
    qa[kc] = frag_cast(w);
  }

  const v8f vzero = {0.f,0.f,0.f,0.f,0.f,0.f,0.f,0.f};
  v8f s[13];
  #pragma unroll
  for (int kt = 0; kt < 13; ++kt) {
    int key = kt * 16 + l16;
    int kr  = key < N_ ? key : N_ - 1;
    const float* kp = kvb + (long)kr * (2 * C_) + h * HD_;
    v8f accs = vzero;
    #pragma unroll
    for (int kc = 0; kc < 2; ++kc) {
      const float* base = kp + kc * 32 + hsel * 16;      // kk = base + 2v (contiguous)
      float4 f0 = *(const float4*)(base + 0);
      float4 f1 = *(const float4*)(base + 4);
      float4 f2 = *(const float4*)(base + 8);
      float4 f3 = *(const float4*)(base + 12);
      v8u w = { pack2(f0.x, f0.y), pack2(f0.z, f0.w),
                pack2(f1.x, f1.y), pack2(f1.z, f1.w),
                pack2(f2.x, f2.y), pack2(f2.z, f2.w),
                pack2(f3.x, f3.y), pack2(f3.z, f3.w) };
      accs = __builtin_amdgcn_wmma_f32_16x16x32_bf16(false, qa[kc], false, frag_cast(w),
                                                     (short)0, accs, false, false);
    }
    const float mask = (key < N_) ? 0.0f : -1e30f;
    #pragma unroll
    for (int r = 0; r < 8; ++r) s[kt][r] = accs[r] * 0.125f + mask;  // hd^-0.5
  }

  // row softmax stats (rows live in 16-lane halves -> xor reduce 1..8)
  float mrow[8], lrow[8];
  #pragma unroll
  for (int r = 0; r < 8; ++r) mrow[r] = -1e30f;
  #pragma unroll
  for (int kt = 0; kt < 13; ++kt)
    #pragma unroll
    for (int r = 0; r < 8; ++r) mrow[r] = fmaxf(mrow[r], s[kt][r]);
  #pragma unroll
  for (int d = 1; d < 16; d <<= 1)
    #pragma unroll
    for (int r = 0; r < 8; ++r) mrow[r] = fmaxf(mrow[r], __shfl_xor(mrow[r], d, 32));
  #pragma unroll
  for (int r = 0; r < 8; ++r) lrow[r] = 0.0f;
  #pragma unroll
  for (int kt = 0; kt < 13; ++kt)
    #pragma unroll
    for (int r = 0; r < 8; ++r) {
      float p = __expf(s[kt][r] - mrow[r]);
      s[kt][r] = p;
      lrow[r] += p;
    }
  #pragma unroll
  for (int d = 1; d < 16; d <<= 1)
    #pragma unroll
    for (int r = 0; r < 8; ++r) lrow[r] += __shfl_xor(lrow[r], d, 32);

  // P @ V : route P through wave-private LDS to build A fragments
  v8f oacc[4];
  #pragma unroll
  for (int dt = 0; dt < 4; ++dt) oacc[dt] = vzero;
  unsigned short* P = &Pbuf[wave][0];
  for (int ch = 0; ch < 7; ++ch) {
    const int kt0 = ch * 2, kt1 = kt0 + 1;
    #pragma unroll
    for (int r = 0; r < 8; ++r) {
      P[(hsel * 8 + r) * 32 + l16]      = f2bf(s[kt0][r]);
      P[(hsel * 8 + r) * 32 + 16 + l16] = (kt1 < 13) ? f2bf(s[kt1][r]) : (unsigned short)0;
    }
    asm volatile("s_wait_dscnt 0" ::: "memory");
    uint4 plo = *(const uint4*)&P[l16 * 32 + hsel * 16];
    uint4 phi = *(const uint4*)&P[l16 * 32 + hsel * 16 + 8];
    v16bf pa = frag_cast(cat4(plo, phi));
    #pragma unroll
    for (int dt = 0; dt < 4; ++dt) {
      const unsigned short* vb = &Vt[dt * 16 + l16][ch * 32 + hsel * 16];
      uint4 vlo = *(const uint4*)(vb + 0);
      uint4 vhi = *(const uint4*)(vb + 8);
      oacc[dt] = __builtin_amdgcn_wmma_f32_16x16x32_bf16(false, pa, false,
                                                         frag_cast(cat4(vlo, vhi)),
                                                         (short)0, oacc[dt], false, false);
    }
    asm volatile("s_wait_dscnt 0" ::: "memory");
  }

  #pragma unroll
  for (int dt = 0; dt < 4; ++dt)
    #pragma unroll
    for (int r = 0; r < 8; ++r) {
      int row = qtile * 16 + hsel * 8 + r;
      if (row < NQ_) {
        o[((long)b * NQ_ + row) * C_ + h * HD_ + dt * 16 + l16] =
            oacc[dt][r] * (1.0f / lrow[r]);
      }
    }
}

// =====================================================================
// LayerNorm over C_=768; one block (256 thr) per row.
// =====================================================================
__global__ __launch_bounds__(256) void ln_kernel(
    const float* __restrict__ x, const float* __restrict__ g,
    const float* __restrict__ bta, float* __restrict__ y)
{
  __shared__ float s1[8], s2[8], sm[2];
  const int row = blockIdx.x;
  const float* xr = x + (long)row * C_;
  float* yr = y + (long)row * C_;
  float lv[3];
  float sum = 0.f, sq = 0.f;
  #pragma unroll
  for (int i = 0; i < 3; ++i) {
    float v = xr[threadIdx.x + i * 256];
    lv[i] = v; sum += v; sq += v * v;
  }
  #pragma unroll
  for (int d = 16; d; d >>= 1) { sum += __shfl_xor(sum, d, 32); sq += __shfl_xor(sq, d, 32); }
  int lane = threadIdx.x & 31, wave = threadIdx.x >> 5;
  if (lane == 0) { s1[wave] = sum; s2[wave] = sq; }
  __syncthreads();
  if (threadIdx.x == 0) {
    float a = 0.f, bsum = 0.f;
    for (int i = 0; i < 8; ++i) { a += s1[i]; bsum += s2[i]; }
    float mean = a * (1.0f / C_);
    float var  = bsum * (1.0f / C_) - mean * mean;
    sm[0] = mean; sm[1] = rsqrtf(var + 1e-6f);
  }
  __syncthreads();
  float mean = sm[0], rstd = sm[1];
  #pragma unroll
  for (int i = 0; i < 3; ++i) {
    int c = threadIdx.x + i * 256;
    yr[c] = (lv[i] - mean) * rstd * g[c] + bta[c];
  }
}

// ---------------- small elementwise / reduction kernels ----------------
__global__ void im2col_kernel(const float* __restrict__ x, float* __restrict__ patches) {
  long idx = (long)blockIdx.x * blockDim.x + threadIdx.x;
  if (idx >= (long)B_ * N_ * 768) return;
  int pd = idx % 768;
  int n  = (idx / 768) % N_;
  int b  = idx / ((long)768 * N_);
  int c = pd >> 8, py = (pd >> 4) & 15, px = pd & 15;
  int i = n / HP_, j = n % HP_;
  patches[idx] = x[(((long)b * 3 + c) * 224 + i * 16 + py) * 224 + (j * 16 + px)];
}

__global__ void copy_tokens_kernel(const float* __restrict__ h, float* __restrict__ Q) {
  long idx = (long)blockIdx.x * blockDim.x + threadIdx.x;
  if (idx >= (long)B_ * N_ * C_) return;
  int c   = idx % C_;
  int tok = (idx / C_) % N_;
  int b   = idx / ((long)N_ * C_);
  Q[((long)b * NQ_ + tok) * C_ + c] = h[idx];
}

__global__ __launch_bounds__(256) void coarse_router_kernel(
    const float* __restrict__ h, const float* __restrict__ rw, const float* __restrict__ rb,
    float* __restrict__ Q, float* __restrict__ gates)
{
  __shared__ float sa[8], sb[8];
  const int b = blockIdx.x / R_;
  const int r = blockIdx.x % R_;
  const int hi = r / 7, hj = r % 7;
  const int t0 = (hi * 2) * HP_ + hj * 2;
  const float* hb = h + (long)b * N_ * C_;
  float d0 = 0.f, d1 = 0.f;
  for (int c = threadIdx.x; c < C_; c += 256) {
    float m = 0.25f * (hb[(long)t0 * C_ + c] + hb[(long)(t0 + 1) * C_ + c] +
                       hb[(long)(t0 + HP_) * C_ + c] + hb[(long)(t0 + HP_ + 1) * C_ + c]);
    Q[((long)b * NQ_ + N_ + r) * C_ + c] = m;
    d0 += m * rw[c * 2 + 0];
    d1 += m * rw[c * 2 + 1];
  }
  #pragma unroll
  for (int d = 16; d; d >>= 1) { d0 += __shfl_xor(d0, d, 32); d1 += __shfl_xor(d1, d, 32); }
  int lane = threadIdx.x & 31, wave = threadIdx.x >> 5;
  if (lane == 0) { sa[wave] = d0; sb[wave] = d1; }
  __syncthreads();
  if (threadIdx.x == 0) {
    float z0 = rb[0], z1 = rb[1];
    for (int i = 0; i < 8; ++i) { z0 += sa[i]; z1 += sb[i]; }
    float mx = fmaxf(z0, z1);
    float e0 = __expf(z0 - mx), e1 = __expf(z1 - mx);
    float inv = 1.0f / (e0 + e1);
    gates[((long)b * R_ + r) * 2 + 0] = e0 * inv;
    gates[((long)b * R_ + r) * 2 + 1] = e1 * inv;
  }
}

__global__ void decompress_kernel(const float* __restrict__ Q, const float* __restrict__ gates,
                                  float* __restrict__ h) {
  long idx = (long)blockIdx.x * blockDim.x + threadIdx.x;
  if (idx >= (long)B_ * N_ * C_) return;
  int c   = idx % C_;
  int tok = (idx / C_) % N_;
  int b   = idx / ((long)N_ * C_);
  int i = tok / HP_, j = tok % HP_;
  int ridx = (i / 2) * 7 + (j / 2);
  float gf = gates[((long)b * R_ + ridx) * 2 + 0];
  float gc = gates[((long)b * R_ + ridx) * 2 + 1];
  h[idx] = gf * Q[((long)b * NQ_ + tok) * C_ + c] +
           gc * Q[((long)b * NQ_ + N_ + ridx) * C_ + c];
}

__global__ __launch_bounds__(256) void pool_kernel(const float* __restrict__ hn,
                                                   float* __restrict__ pooled) {
  int b = blockIdx.x;
  for (int c = threadIdx.x; c < C_; c += 256) {
    float s = 0.f;
    for (int tkn = 0; tkn < N_; ++tkn) s += hn[((long)b * N_ + tkn) * C_ + c];
    pooled[(long)b * C_ + c] = s * (1.0f / N_);
  }
}

// =====================================================================
// Host orchestration
// =====================================================================
extern "C" void kernel_launch(void* const* d_in, const int* in_sizes, int n_in,
                              void* d_out, int out_size, void* d_ws, size_t ws_size,
                              hipStream_t stream) {
  (void)in_sizes; (void)n_in; (void)out_size;
  const float* x       = (const float*)d_in[0];
  const float* patch_w = (const float*)d_in[1];
  const float* patch_b = (const float*)d_in[2];
  const float* pos     = (const float*)d_in[3];
  const int nb = 4 + 16 * DEPTH_;
  const float* norm_g = (const float*)d_in[nb + 0];
  const float* norm_b = (const float*)d_in[nb + 1];
  const float* head_w = (const float*)d_in[nb + 2];
  const float* head_b = (const float*)d_in[nb + 3];

  // workspace layout (floats)
  float* ws = (float*)d_ws;
  size_t oH    = 0;                                  // MLP hidden / im2col patches
  size_t oh    = oH    + (size_t)B_ * NQ_ * FF_;
  size_t oQ    = oh    + (size_t)B_ * N_  * C_;
  size_t oqn   = oQ    + (size_t)B_ * NQ_ * C_;
  size_t oqh   = oqn   + (size_t)B_ * NQ_ * C_;
  size_t okv   = oqh   + (size_t)B_ * NQ_ * C_;
  size_t oo    = okv   + (size_t)B_ * N_  * 2 * C_;
  size_t opool = oo    + (size_t)B_ * NQ_ * C_;
  size_t ogate = opool + (size_t)B_ * C_;
  size_t need  = ogate + (size_t)B_ * R_ * 2;
  if (ws_size < need * sizeof(float)) return;
  float* Hbuf  = ws + oH;
  float* hbuf  = ws + oh;
  float* Qbuf  = ws + oQ;
  float* qn    = ws + oqn;
  float* qh    = ws + oqh;     // also reused for final LN output
  float* kvbuf = ws + okv;
  float* obuf  = ws + oo;
  float* pooled= ws + opool;
  float* gates = ws + ogate;

  const long toksz = (long)B_ * N_ * C_;

  // patch embedding: im2col + GEMM (pos_embed as stride-0 residual)
  im2col_kernel<<<dim3((unsigned)((toksz + 255) / 256)), dim3(256), 0, stream>>>(x, Hbuf);
  gemm_kernel<<<dim3(2, 6, B_), dim3(256), 0, stream>>>(
      Hbuf, patch_w, patch_b, pos, hbuf,
      N_, 768, C_, (long)N_ * 768, (long)N_ * C_, 0L, 0);

  for (int L = 0; L < DEPTH_; ++L) {
    const int bs = 4 + L * 16;
    const float* n1g = (const float*)d_in[bs + 0];
    const float* n1b = (const float*)d_in[bs + 1];
    const float* qw  = (const float*)d_in[bs + 2];
    const float* qb  = (const float*)d_in[bs + 3];
    const float* kvw = (const float*)d_in[bs + 4];
    const float* kvb = (const float*)d_in[bs + 5];
    const float* pw  = (const float*)d_in[bs + 6];
    const float* pb  = (const float*)d_in[bs + 7];
    const float* n2g = (const float*)d_in[bs + 8];
    const float* n2b = (const float*)d_in[bs + 9];
    const float* f1w = (const float*)d_in[bs + 10];
    const float* f1b = (const float*)d_in[bs + 11];
    const float* f2w = (const float*)d_in[bs + 12];
    const float* f2b = (const float*)d_in[bs + 13];
    const float* rww = (const float*)d_in[bs + 14];
    const float* rbb = (const float*)d_in[bs + 15];

    copy_tokens_kernel<<<dim3((unsigned)((toksz + 255) / 256)), dim3(256), 0, stream>>>(hbuf, Qbuf);
    coarse_router_kernel<<<dim3(B_ * R_), dim3(256), 0, stream>>>(hbuf, rww, rbb, Qbuf, gates);
    ln_kernel<<<dim3(B_ * NQ_), dim3(256), 0, stream>>>(Qbuf, n1g, n1b, qn);

    // q = qn @ qw + qb
    gemm_kernel<<<dim3(2, 6, B_), dim3(256), 0, stream>>>(
        qn, qw, qb, nullptr, qh,
        NQ_, C_, C_, (long)NQ_ * C_, (long)NQ_ * C_, 0L, 0);
    // kv = xn @ kvw + kvb   (xn = first 196 rows of qn per batch)
    gemm_kernel<<<dim3(2, 12, B_), dim3(256), 0, stream>>>(
        qn, kvw, kvb, nullptr, kvbuf,
        N_, C_, 2 * C_, (long)NQ_ * C_, (long)N_ * 2 * C_, 0L, 0);

    attn_kernel<<<dim3(B_ * NH_, 2), dim3(256), 0, stream>>>(qh, kvbuf, obuf);

    // q += o @ pw + pb (in-place residual)
    gemm_kernel<<<dim3(2, 6, B_), dim3(256), 0, stream>>>(
        obuf, pw, pb, Qbuf, Qbuf,
        NQ_, C_, C_, (long)NQ_ * C_, (long)NQ_ * C_, (long)NQ_ * C_, 0);

    ln_kernel<<<dim3(B_ * NQ_), dim3(256), 0, stream>>>(Qbuf, n2g, n2b, qn);
    // H = gelu(qn @ f1w + f1b)
    gemm_kernel<<<dim3(2, 24, B_), dim3(256), 0, stream>>>(
        qn, f1w, f1b, nullptr, Hbuf,
        NQ_, C_, FF_, (long)NQ_ * C_, (long)NQ_ * FF_, 0L, 1);
    // q += H @ f2w + f2b
    gemm_kernel<<<dim3(2, 6, B_), dim3(256), 0, stream>>>(
        Hbuf, f2w, f2b, Qbuf, Qbuf,
        NQ_, FF_, C_, (long)NQ_ * FF_, (long)NQ_ * C_, (long)NQ_ * C_, 0);

    decompress_kernel<<<dim3((unsigned)((toksz + 255) / 256)), dim3(256), 0, stream>>>(
        Qbuf, gates, hbuf);
  }

  // final LN -> mean pool -> head
  ln_kernel<<<dim3(B_ * N_), dim3(256), 0, stream>>>(hbuf, norm_g, norm_b, qh);
  pool_kernel<<<dim3(B_), dim3(256), 0, stream>>>(qh, pooled);
  gemm_kernel<<<dim3(1, 8, 1), dim3(256), 0, stream>>>(
      pooled, head_w, head_b, nullptr, (float*)d_out,
      B_, C_, NCLS_, 0L, 0L, 0L, 0);
}
